// CrossAttentionLayer_40407052320817
// MI455X (gfx1250) — compile-verified, matrix operand order checked by reading, not run on hardware
//
#include <hip/hip_runtime.h>
#include <hip/hip_bf16.h>

typedef __attribute__((ext_vector_type(16))) _Float16 v16h;
typedef __attribute__((ext_vector_type(8)))  _Float16 v8h;
typedef __attribute__((ext_vector_type(8)))  float    v8f;

constexpr int Bc = 32, Sc = 512, Dc = 1024, Hc = 16, DHc = 64;

__device__ __forceinline__ v8f wmma_f16(v16h a, v16h b, v8f c) {
  return __builtin_amdgcn_wmma_f32_16x16x32_f16(false, a, false, b, (short)0, c,
                                                false, false);
}

// Load one 16x32 A/B fragment (ISA 7.12.2 layout) from a row-major tile:
// lane&15 = row (or column for B from a [N,K] buffer), halves = K pairs.
__device__ __forceinline__ v16h load_frag(const _Float16* tile_base, int ld, int k0) {
  int lane = threadIdx.x & 31;
  int r  = lane & 15;
  int hf = lane >> 4;
  const _Float16* p = tile_base + (size_t)r * ld + k0 + hf * 8;
  v8h lo = *(const v8h*)p;         // k = k0 + hf*8 .. +7
  v8h hi = *(const v8h*)(p + 16);  // k = k0 + 16 + hf*8 .. +7
  v16h out;
#pragma unroll
  for (int i = 0; i < 8; ++i) { out[i] = lo[i]; out[i + 8] = hi[i]; }
  return out;
}

// ---- CDNA5 async copy: 16B per lane, global -> LDS, tracked by ASYNCcnt -------
__device__ __forceinline__ void async_b128(unsigned lds_off, const _Float16* g) {
  asm volatile("global_load_async_to_lds_b128 %0, %1, off"
               :: "v"(lds_off), "v"(g) : "memory");
}
__device__ __forceinline__ void wait_async_le4() {
  asm volatile("s_wait_asynccnt 0x4" ::: "memory");
}
__device__ __forceinline__ void wait_async_0() {
  asm volatile("s_wait_asynccnt 0x0" ::: "memory");
}

// ---- fp32 weight [K,N] -> f16 transposed [N,K] --------------------------------
__global__ void convert_w(const float* __restrict__ W, _Float16* __restrict__ Wt) {
  int idx = blockIdx.x * 256 + threadIdx.x;   // over K*N = 1M
  int k = idx >> 10, n = idx & 1023;
  Wt[(size_t)n * Dc + k] = (_Float16)W[idx];
}

// ---- LayerNorm: fp32 row -> f16 normalized (+ optional fp32 copy) -------------
__global__ void ln_kernel(const float* __restrict__ x, const float* __restrict__ g,
                          const float* __restrict__ bb, _Float16* __restrict__ yh,
                          float* __restrict__ yf) {
  __shared__ float red[256];
  int row = blockIdx.x, tid = threadIdx.x;
  const float* xr = x + (size_t)row * Dc;
  float v[4], s = 0.f;
#pragma unroll
  for (int i = 0; i < 4; ++i) { v[i] = xr[tid + i * 256]; s += v[i]; }
  red[tid] = s; __syncthreads();
  for (int off = 128; off > 0; off >>= 1) {
    if (tid < off) red[tid] += red[tid + off];
    __syncthreads();
  }
  float mu = red[0] * (1.0f / Dc);
  __syncthreads();
  s = 0.f;
#pragma unroll
  for (int i = 0; i < 4; ++i) { float d = v[i] - mu; s += d * d; }
  red[tid] = s; __syncthreads();
  for (int off = 128; off > 0; off >>= 1) {
    if (tid < off) red[tid] += red[tid + off];
    __syncthreads();
  }
  float rstd = rsqrtf(red[0] * (1.0f / Dc) + 1e-5f);
#pragma unroll
  for (int i = 0; i < 4; ++i) {
    int c = tid + i * 256;
    float y = (v[i] - mu) * rstd * g[c] + bb[c];
    yh[(size_t)row * Dc + c] = (_Float16)y;
    if (yf) yf[(size_t)row * Dc + c] = y;
  }
}

// ---- WMMA GEMM, block-cooperative with double-buffered async LDS staging ------
// Block = 4 waves -> 64x64 C tile; per k-step stage A(64x32) + W(64x32) in LDS.
// mode 0: f16 row-major out; mode 1: f16 out transposed per-(b,h) [dh][s] (V);
// mode 2: fp32 out + residual.
__global__ void gemm_kernel(const _Float16* __restrict__ A,
                            const _Float16* __restrict__ Wt,
                            const float* __restrict__ bias, int mode,
                            _Float16* __restrict__ outH, float* __restrict__ outF,
                            const float* __restrict__ resid) {
  const int N = Dc, K = Dc;
  __shared__ __attribute__((aligned(16))) _Float16 As[2][64 * 32];
  __shared__ __attribute__((aligned(16))) _Float16 Ws[2][64 * 32];
  int t = threadIdx.x;            // 0..127
  int w = t >> 5;                 // wave in block
  int lane = t & 31;
  int m0 = (blockIdx.x >> 4) * 64;   // 256 M blocks
  int n0 = (blockIdx.x & 15) * 64;   // 16 N blocks

  // Cooperative staging: tile = 64 rows x 32 halves = 256 x 16B chunks,
  // chunk c -> row c>>2, 16B sub-chunk c&3; LDS addr = tile + c*16B.
  const int c0 = t, c1 = t + 128;
  const _Float16* Abase = A + (size_t)m0 * K;
  const _Float16* Wbase = Wt + (size_t)n0 * K;
  unsigned aoff[2], woff[2];
  aoff[0] = (unsigned)(uintptr_t)&As[0][0];
  aoff[1] = (unsigned)(uintptr_t)&As[1][0];
  woff[0] = (unsigned)(uintptr_t)&Ws[0][0];
  woff[1] = (unsigned)(uintptr_t)&Ws[1][0];

  auto stage = [&](int buf, int k0) {
    async_b128(aoff[buf] + c0 * 16, Abase + (size_t)(c0 >> 2) * K + k0 + (c0 & 3) * 8);
    async_b128(aoff[buf] + c1 * 16, Abase + (size_t)(c1 >> 2) * K + k0 + (c1 & 3) * 8);
    async_b128(woff[buf] + c0 * 16, Wbase + (size_t)(c0 >> 2) * K + k0 + (c0 & 3) * 8);
    async_b128(woff[buf] + c1 * 16, Wbase + (size_t)(c1 >> 2) * K + k0 + (c1 & 3) * 8);
  };

  v8f acc[4] = {};
  stage(0, 0);
  const int KSTEPS = K / 32;      // 32
  for (int i = 0; i < KSTEPS; ++i) {
    if (i + 1 < KSTEPS) {
      stage((i + 1) & 1, (i + 1) * 32);  // prefetch next buffer
      wait_async_le4();                  // retire current buffer only
    } else {
      wait_async_0();
    }
    __syncthreads();                     // all waves' staging visible
    int buf = i & 1;
    v16h af = load_frag(&As[buf][w * 16 * 32], 32, 0);
#pragma unroll
    for (int tt = 0; tt < 4; ++tt) {
      v16h bf = load_frag(&Ws[buf][tt * 16 * 32], 32, 0);
      acc[tt] = wmma_f16(af, bf, acc[tt]);
    }
    __syncthreads();                     // protect buffer before re-issue
  }

  int mw = m0 + w * 16;
  int cn = lane & 15, hf = lane >> 4;
#pragma unroll
  for (int tt = 0; tt < 4; ++tt) {
    int n = n0 + tt * 16 + cn;
    float bvv = bias[n];
#pragma unroll
    for (int i = 0; i < 8; ++i) {
      int m = mw + hf * 8 + i;
      float val = acc[tt][i] + bvv;
      if (mode == 0) {
        outH[(size_t)m * N + n] = (_Float16)val;
      } else if (mode == 1) {
        int bidx = m >> 9, s = m & 511;
        int hh = n >> 6, dh = n & 63;
        outH[(((size_t)bidx * Hc + hh) * DHc + dh) * Sc + s] = (_Float16)val;
      } else {
        outF[(size_t)m * N + n] = val + resid[(size_t)m * N + n];
      }
    }
  }
}

// ---- Flash attention: one wave per (b, h, 16-query tile) ----------------------
__global__ void attn_kernel(const _Float16* __restrict__ Qh,
                            const _Float16* __restrict__ Kh,
                            const _Float16* __restrict__ Vt,
                            const int* __restrict__ mask,
                            _Float16* __restrict__ ctx) {
  __shared__ __attribute__((aligned(16))) _Float16 lds[4][16 * 32];
  int wv = threadIdx.x >> 5;
  int wave = blockIdx.x * 4 + wv;
  int qt = wave & 31;          // S/16 = 32 query tiles
  int h  = (wave >> 5) & 15;
  int b  = wave >> 9;
  int lane = threadIdx.x & 31;
  int r = lane & 15, hf = lane >> 4;

  const _Float16* qbase = Qh + (size_t)(b * Sc + qt * 16) * Dc + h * DHc;
  v16h qa0 = load_frag(qbase, Dc, 0);
  v16h qa1 = load_frag(qbase, Dc, 32);

  v8f acc[4] = {};
  float mrow[8], lrow[8];
#pragma unroll
  for (int i = 0; i < 8; ++i) { mrow[i] = -3.0e38f; lrow[i] = 0.f; }

  const int* mptr = mask + b * Sc;
  const float scale = 0.125f;  // 1/sqrt(64)

  for (int j0 = 0; j0 < Sc; j0 += 32) {
    const _Float16* kb0 = Kh + (size_t)(b * Sc + j0) * Dc + h * DHc;
    const _Float16* kb1 = kb0 + (size_t)16 * Dc;
    v8f s0 = {}, s1 = {};
    s0 = wmma_f16(qa0, load_frag(kb0, Dc, 0),  s0);
    s0 = wmma_f16(qa1, load_frag(kb0, Dc, 32), s0);
    s1 = wmma_f16(qa0, load_frag(kb1, Dc, 0),  s1);
    s1 = wmma_f16(qa1, load_frag(kb1, Dc, 32), s1);

    int mk0 = mptr[j0 + r];
    int mk1 = mptr[j0 + 16 + r];
#pragma unroll
    for (int i = 0; i < 8; ++i) {
      s0[i] = (mk0 == 0) ? -1.0e9f : s0[i] * scale;
      s1[i] = (mk1 == 0) ? -1.0e9f : s1[i] * scale;
    }
    // row-wise max across 16-lane column group
    float nm[8];
#pragma unroll
    for (int i = 0; i < 8; ++i) nm[i] = fmaxf(s0[i], s1[i]);
#pragma unroll
    for (int off = 1; off < 16; off <<= 1)
#pragma unroll
      for (int i = 0; i < 8; ++i) nm[i] = fmaxf(nm[i], __shfl_xor(nm[i], off, 32));

    float p0[8], p1[8], ts[8], corr[8];
#pragma unroll
    for (int i = 0; i < 8; ++i) {
      float mn = fmaxf(mrow[i], nm[i]);
      corr[i] = __expf(mrow[i] - mn);
      mrow[i] = mn;
      p0[i] = __expf(s0[i] - mn);
      p1[i] = __expf(s1[i] - mn);
      ts[i] = p0[i] + p1[i];
    }
#pragma unroll
    for (int off = 1; off < 16; off <<= 1)
#pragma unroll
      for (int i = 0; i < 8; ++i) ts[i] += __shfl_xor(ts[i], off, 32);
#pragma unroll
    for (int i = 0; i < 8; ++i) {
      lrow[i] = lrow[i] * corr[i] + ts[i];
#pragma unroll
      for (int t = 0; t < 4; ++t) acc[t][i] *= corr[i];
    }
    // stage probs through LDS to reshape C-layout -> A-fragment layout
    _Float16* L = lds[wv];
#pragma unroll
    for (int i = 0; i < 8; ++i) {
      L[(hf * 8 + i) * 32 + r]      = (_Float16)p0[i];
      L[(hf * 8 + i) * 32 + 16 + r] = (_Float16)p1[i];
    }
    v8h lo = *(const v8h*)&L[r * 32 + hf * 8];
    v8h hi = *(const v8h*)&L[r * 32 + 16 + hf * 8];
    v16h pa;
#pragma unroll
    for (int i = 0; i < 8; ++i) { pa[i] = lo[i]; pa[i + 8] = hi[i]; }
#pragma unroll
    for (int t = 0; t < 4; ++t) {
      const _Float16* vb = Vt + (((size_t)b * Hc + h) * DHc + t * 16) * Sc;
      acc[t] = wmma_f16(pa, load_frag(vb, Sc, j0), acc[t]);
    }
  }
#pragma unroll
  for (int t = 0; t < 4; ++t) {
    int n = h * DHc + t * 16 + r;
#pragma unroll
    for (int i = 0; i < 8; ++i) {
      int m = b * Sc + qt * 16 + hf * 8 + i;
      ctx[(size_t)m * Dc + n] = (_Float16)(acc[t][i] / lrow[i]);
    }
  }
}

extern "C" void kernel_launch(void* const* d_in, const int* in_sizes, int n_in,
                              void* d_out, int out_size, void* d_ws, size_t ws_size,
                              hipStream_t stream) {
  (void)in_sizes; (void)n_in; (void)out_size; (void)ws_size;
  const float* qk   = (const float*)d_in[0];
  const float* v    = (const float*)d_in[1];
  const int*   mask = (const int*)d_in[2];
  const float* qk_g = (const float*)d_in[3];
  const float* qk_b = (const float*)d_in[4];
  const float* v_g  = (const float*)d_in[5];
  const float* v_b  = (const float*)d_in[6];
  const float* Wq = (const float*)d_in[7];
  const float* bq = (const float*)d_in[8];
  const float* Wk = (const float*)d_in[9];
  const float* bk = (const float*)d_in[10];
  const float* Wv = (const float*)d_in[11];
  const float* bv = (const float*)d_in[12];
  const float* Wo = (const float*)d_in[13];
  const float* bo = (const float*)d_in[14];
  float* out = (float*)d_out;

  char* ws = (char*)d_ws;
  size_t off = 0;
  auto carve = [&](size_t bytes) -> void* {
    void* p = ws + off;
    off += (bytes + 255) & ~size_t(255);
    return p;
  };
  const size_t tok = (size_t)Bc * Sc;                 // 16384 rows
  _Float16* qk_nh = (_Float16*)carve(tok * Dc * 2);
  _Float16* v_nh  = (_Float16*)carve(tok * Dc * 2);
  float*    v_nf  = (float*)   carve(tok * Dc * 4);
  _Float16* Qh    = (_Float16*)carve(tok * Dc * 2);
  _Float16* Khb   = (_Float16*)carve(tok * Dc * 2);
  _Float16* Vtt   = (_Float16*)carve(tok * Dc * 2);
  _Float16* ctxh  = (_Float16*)carve(tok * Dc * 2);
  _Float16* Wqt   = (_Float16*)carve((size_t)Dc * Dc * 2);
  _Float16* Wkt   = (_Float16*)carve((size_t)Dc * Dc * 2);
  _Float16* Wvt   = (_Float16*)carve((size_t)Dc * Dc * 2);
  _Float16* Wot   = (_Float16*)carve((size_t)Dc * Dc * 2);

  convert_w<<<4096, 256, 0, stream>>>(Wq, Wqt);
  convert_w<<<4096, 256, 0, stream>>>(Wk, Wkt);
  convert_w<<<4096, 256, 0, stream>>>(Wv, Wvt);
  convert_w<<<4096, 256, 0, stream>>>(Wo, Wot);

  ln_kernel<<<16384, 256, 0, stream>>>(qk, qk_g, qk_b, qk_nh, nullptr);
  ln_kernel<<<16384, 256, 0, stream>>>(v,  v_g,  v_b,  v_nh,  v_nf);

  // Q, K projections (f16 out row-major)
  gemm_kernel<<<4096, 128, 0, stream>>>(qk_nh, Wqt, bq, 0, Qh, nullptr, nullptr);
  gemm_kernel<<<4096, 128, 0, stream>>>(qk_nh, Wkt, bk, 0, Khb, nullptr, nullptr);
  // V projection, stored transposed per (b,h): [dh][s]
  gemm_kernel<<<4096, 128, 0, stream>>>(v_nh, Wvt, bv, 1, Vtt, nullptr, nullptr);

  // attention: B*H*(S/16) = 16384 waves
  attn_kernel<<<4096, 128, 0, stream>>>(Qh, Khb, Vtt, mask, ctxh);

  // output projection + bias + residual (fp32 out)
  gemm_kernel<<<4096, 128, 0, stream>>>(ctxh, Wot, bo, 2, nullptr, out, v_nf);
}
